// MCA_40510131536220
// MI455X (gfx1250) — matrix operand order repaired and checked by reading, not compiled, not run on hardware
//
#include <hip/hip_runtime.h>
#include <hip/hip_bf16.h>

typedef __bf16 v16bf __attribute__((ext_vector_type(16)));
typedef float  v8f   __attribute__((ext_vector_type(8)));

#define BATCH 16
#define C     512
#define D     4096
#define SCALE 0.015625f   // 1/sqrt(4096)

#define APITCH 72   // LDS pitch (bf16 elems) for 64-deep K slices: 144B rows,
                    // row-base bank = (r*36)%64 -> distinct for r=0..15
#define BPITCH 34   // k_out transposed tile pitch: 68B rows, (n*17)%64 distinct

// ---------------------------------------------------------------------------
// WMMA fragment loaders (wave32, v_wmma_f32_16x16x32_bf16 layouts per ISA 7.12.2)
// A (16x32 MxK): lane r=lane&15, h=lane>>4: e[0..7]=K+8h+0..7, e[8..15]=K+16+8h+0..7
// B (32x16 KxN), N-major source: lane n=lane&15: e[0..7]=K+8h+0..7, e[8..15]=K+16+8h+0..7
// ---------------------------------------------------------------------------

__device__ inline v16bf frag_a_bf16(const __bf16* __restrict__ tile, int ld, int k0) {
    int lane = threadIdx.x & 31;
    int r = lane & 15, h = lane >> 4;
    const __bf16* p = tile + (size_t)r * ld + k0 + 8 * h;
    v16bf a;
#pragma unroll
    for (int j = 0; j < 8; ++j) a[j] = p[j];
#pragma unroll
    for (int j = 0; j < 8; ++j) a[8 + j] = p[16 + j];
    return a;
}

__device__ inline v16bf frag_b_nt_bf16(const __bf16* __restrict__ tile, int ld, int k0) {
    int lane = threadIdx.x & 31;
    int n = lane & 15, h = lane >> 4;
    const __bf16* p = tile + (size_t)n * ld + k0 + 8 * h;
    v16bf b;
#pragma unroll
    for (int j = 0; j < 8; ++j) b[j] = p[j];
#pragma unroll
    for (int j = 0; j < 8; ++j) b[8 + j] = p[16 + j];
    return b;
}

__device__ inline v8f wmma_bf16(v16bf a, v16bf b, v8f c) {
    return __builtin_amdgcn_wmma_f32_16x16x32_bf16(false, a, false, b, (short)0, c,
                                                   false, false);
}

// ---------------------------------------------------------------------------
// Kernel 0: f32 -> bf16 bulk convert (one-shot, HBM-rate)
// ---------------------------------------------------------------------------
__global__ __launch_bounds__(256) void k_convert(const float* __restrict__ src,
                                                 __bf16* __restrict__ dst) {
    size_t i = ((size_t)blockIdx.x * 256 + threadIdx.x) * 8;
    float4 a = *(const float4*)(src + i);
    float4 b = *(const float4*)(src + i + 4);
    __bf16* o = dst + i;
    o[0] = (__bf16)a.x; o[1] = (__bf16)a.y; o[2] = (__bf16)a.z; o[3] = (__bf16)a.w;
    o[4] = (__bf16)b.x; o[5] = (__bf16)b.y; o[6] = (__bf16)b.z; o[7] = (__bf16)b.w;
}

// ---------------------------------------------------------------------------
// Async stage: 128 rows x 64 bf16 (16B segments) -> LDS (pitch APITCH), via
// GLOBAL_LOAD_ASYNC_TO_LDS_B128 (ASYNCcnt-tracked, no VGPR round trip).
// 256 threads x 4 segments = 1024 x 16B = 16KB.
// ---------------------------------------------------------------------------
__device__ inline void stage_async_128x64(unsigned int lds_base_bytes,
                                          const __bf16* __restrict__ gsrc,
                                          int ld, int k0) {
    int t = threadIdx.x;
#pragma unroll
    for (int it = 0; it < 4; ++it) {
        int seg = t + 256 * it;        // 0..1023
        int row = seg >> 3;
        int s   = seg & 7;             // 16B segment within the 128B row
        unsigned long long g =
            (unsigned long long)(uintptr_t)(gsrc + (size_t)row * ld + k0 + s * 8);
        unsigned int l = lds_base_bytes + (unsigned int)(row * APITCH + s * 8) * 2u;
        asm volatile("global_load_async_to_lds_b128 %0, %1, off"
                     :: "v"(l), "v"(g) : "memory");
    }
}

// ---------------------------------------------------------------------------
// Kernel 1: S[b] = (q . k^T) * scale, bf16 WMMA, NT GEMM, M=N=512, K=4096.
// 128x128 block tile, 8 waves of 32x64. Double-buffered async LDS staging,
// K-chunk = 64 (two 32-deep WMMA sub-steps per chunk).
// ---------------------------------------------------------------------------
__global__ __launch_bounds__(256) void k_scores(const __bf16* __restrict__ qb,
                                                const __bf16* __restrict__ kb,
                                                float* __restrict__ S) {
    const int b  = blockIdx.z;
    const int m0 = blockIdx.y * 128;
    const int n0 = blockIdx.x * 128;
    const int wave = threadIdx.x >> 5;
    const int wm = (wave & 3) * 32;   // 4 waves along M
    const int wn = (wave >> 2) * 64;  // 2 waves along N

    const __bf16* q = qb + (size_t)b * C * D + (size_t)m0 * D;
    const __bf16* k = kb + (size_t)b * C * D + (size_t)n0 * D;

    __shared__ __bf16 lA[2][128 * APITCH];
    __shared__ __bf16 lB[2][128 * APITCH];
    const unsigned int baseA0 = (unsigned int)(uintptr_t)&lA[0][0];
    const unsigned int baseA1 = (unsigned int)(uintptr_t)&lA[1][0];
    const unsigned int baseB0 = (unsigned int)(uintptr_t)&lB[0][0];
    const unsigned int baseB1 = (unsigned int)(uintptr_t)&lB[1][0];

    v8f acc[2][4] = {};

    // prologue: chunk 0 into buffer 0
    stage_async_128x64(baseA0, q, D, 0);
    stage_async_128x64(baseB0, k, D, 0);

    int p = 0;
    for (int k0 = 0; k0 < D; k0 += 64) {
        asm volatile("s_wait_asynccnt 0" ::: "memory");
        __syncthreads();
        if (k0 + 64 < D) {
            stage_async_128x64(p ? baseA0 : baseA1, q, D, k0 + 64);
            stage_async_128x64(p ? baseB0 : baseB1, k, D, k0 + 64);
        }
        const __bf16* Ab = lA[p];
        const __bf16* Bb = lB[p];
#pragma unroll
        for (int kk = 0; kk < 64; kk += 32) {
            v16bf af[2], bf[4];
#pragma unroll
            for (int i = 0; i < 2; ++i)
                af[i] = frag_a_bf16(Ab + (wm + 16 * i) * APITCH, APITCH, kk);
#pragma unroll
            for (int j = 0; j < 4; ++j)
                bf[j] = frag_b_nt_bf16(Bb + (wn + 16 * j) * APITCH, APITCH, kk);
#pragma unroll
            for (int i = 0; i < 2; ++i)
#pragma unroll
                for (int j = 0; j < 4; ++j)
                    acc[i][j] = wmma_bf16(af[i], bf[j], acc[i][j]);
        }
        p ^= 1;
    }

    const int lane  = threadIdx.x & 31;
    const int nlane = lane & 15;
    const int mh    = (lane >> 4) * 8;
#pragma unroll
    for (int i = 0; i < 2; ++i)
#pragma unroll
        for (int j = 0; j < 4; ++j)
#pragma unroll
            for (int r = 0; r < 8; ++r) {
                int row = m0 + wm + 16 * i + r + mh;
                int col = n0 + wn + 16 * j + nlane;
                S[((size_t)b * C + row) * C + col] = acc[i][j][r] * SCALE;
            }
}

// ---------------------------------------------------------------------------
// Kernel 2: row softmax of S -> Pa (bf16). One block per (b,c) row of 512.
// ---------------------------------------------------------------------------
__global__ __launch_bounds__(256) void k_softmax_row(const float* __restrict__ S,
                                                     __bf16* __restrict__ Pa) {
    const size_t row = blockIdx.x;               // b*C + c
    const float* s = S + row * C;
    const int t = threadIdx.x;
    float v0 = s[t], v1 = s[t + 256];

    __shared__ float red[256];
    red[t] = fmaxf(v0, v1);
    __syncthreads();
    for (int off = 128; off > 0; off >>= 1) {
        if (t < off) red[t] = fmaxf(red[t], red[t + off]);
        __syncthreads();
    }
    const float m = red[0];
    __syncthreads();

    float e0 = __expf(v0 - m), e1 = __expf(v1 - m);
    red[t] = e0 + e1;
    __syncthreads();
    for (int off = 128; off > 0; off >>= 1) {
        if (t < off) red[t] += red[t + off];
        __syncthreads();
    }
    const float inv = 1.0f / red[0];

    Pa[row * C + t]       = (__bf16)(e0 * inv);
    Pa[row * C + t + 256] = (__bf16)(e1 * inv);
}

// ---------------------------------------------------------------------------
// Kernel 3: column softmax of S -> Pb (bf16), Pb[b][e][c] = softmax_c(S[b][c][e]).
// One block per (b,e); strided column reads are served from L2 (S = 16MB).
// ---------------------------------------------------------------------------
__global__ __launch_bounds__(256) void k_softmax_col(const float* __restrict__ S,
                                                     __bf16* __restrict__ Pb) {
    const int be = blockIdx.x;
    const int b  = be >> 9;
    const int e  = be & 511;
    const float* s = S + (size_t)b * C * C + e;   // element c at s[c*C]
    const int t = threadIdx.x;
    float v0 = s[(size_t)t * C], v1 = s[(size_t)(t + 256) * C];

    __shared__ float red[256];
    red[t] = fmaxf(v0, v1);
    __syncthreads();
    for (int off = 128; off > 0; off >>= 1) {
        if (t < off) red[t] = fmaxf(red[t], red[t + off]);
        __syncthreads();
    }
    const float m = red[0];
    __syncthreads();

    float e0 = __expf(v0 - m), e1 = __expf(v1 - m);
    red[t] = e0 + e1;
    __syncthreads();
    for (int off = 128; off > 0; off >>= 1) {
        if (t < off) red[t] += red[t + off];
        __syncthreads();
    }
    const float inv = 1.0f / red[0];

    __bf16* dst = Pb + ((size_t)b * C + e) * C;
    dst[t]       = (__bf16)(e0 * inv);
    dst[t + 256] = (__bf16)(e1 * inv);
}

// ---------------------------------------------------------------------------
// Kernel 4: out[b] = P[b] (bf16 512x512) . V[b] (bf16 512x4096)  (NN GEMM)
// A fragments straight from global (K-contiguous). B staged through LDS with
// transpose-on-write (vectorized 16B bf16 loads, b16 scatter to pitch-34 rows).
// ---------------------------------------------------------------------------
__global__ __launch_bounds__(256) void k_out(const __bf16* __restrict__ P,
                                             const __bf16* __restrict__ V,
                                             float* __restrict__ out) {
    const int b  = blockIdx.z;
    const int m0 = blockIdx.y * 128;
    const int n0 = blockIdx.x * 128;
    const int wave = threadIdx.x >> 5;
    const int wm = (wave & 3) * 32;
    const int wn = (wave >> 2) * 64;

    const __bf16* A  = P + (size_t)b * C * C;
    const __bf16* Vb = V + (size_t)b * C * D;
    float*        Ob = out + (size_t)b * C * D;

    __shared__ __bf16 ldsBT[128 * BPITCH];   // [n][kk], kk in 0..31

    v8f acc[2][4] = {};

    for (int k0 = 0; k0 < C; k0 += 32) {
        __syncthreads();   // previous iteration's fragment reads done
#pragma unroll
        for (int pass = 0; pass < 2; ++pass) {
            int kk = (threadIdx.x >> 4) + pass * 16;     // 0..31
            int n8 = (threadIdx.x & 15) * 8;             // 16B-aligned col group
            const __bf16* g = Vb + (size_t)(k0 + kk) * D + n0 + n8;
#pragma unroll
            for (int i = 0; i < 8; ++i)
                ldsBT[(n8 + i) * BPITCH + kk] = g[i];
        }
        __syncthreads();

        v16bf af[2], bf[4];
#pragma unroll
        for (int i = 0; i < 2; ++i)
            af[i] = frag_a_bf16(A + (size_t)(m0 + wm + 16 * i) * C, C, k0);
#pragma unroll
        for (int j = 0; j < 4; ++j)
            bf[j] = frag_b_nt_bf16(&ldsBT[(wn + 16 * j) * BPITCH], BPITCH, 0);
#pragma unroll
        for (int i = 0; i < 2; ++i)
#pragma unroll
            for (int j = 0; j < 4; ++j)
                acc[i][j] = wmma_bf16(af[i], bf[j], acc[i][j]);
    }

    const int lane  = threadIdx.x & 31;
    const int nlane = lane & 15;
    const int mh    = (lane >> 4) * 8;
#pragma unroll
    for (int i = 0; i < 2; ++i)
#pragma unroll
        for (int j = 0; j < 4; ++j)
#pragma unroll
            for (int r = 0; r < 8; ++r) {
                int row = m0 + wm + 16 * i + r + mh;
                int col = n0 + wn + 16 * j + nlane;
                Ob[(size_t)row * D + col] = acc[i][j][r];
            }
}

// ---------------------------------------------------------------------------
extern "C" void kernel_launch(void* const* d_in, const int* in_sizes, int n_in,
                              void* d_out, int out_size, void* d_ws, size_t ws_size,
                              hipStream_t stream) {
    const float* x1 = (const float*)d_in[0];
    const float* x2 = (const float*)d_in[1];
    float* out = (float*)d_out;

    const size_t NE = (size_t)BATCH * C * D;   // 33.5M elems per tensor

    // workspace: qb (bf16 64MB) | kb (bf16 64MB) | S (f32 16MB) | Pa | Pb (bf16 8MB ea)
    __bf16* qb = (__bf16*)d_ws;
    __bf16* kb = qb + NE;
    float*  S  = (float*)(kb + NE);
    __bf16* Pa = (__bf16*)(S + (size_t)BATCH * C * C);
    __bf16* Pb = Pa + (size_t)BATCH * C * C;

    const int convBlocks = (int)(NE / (256 * 8));
    k_convert<<<convBlocks, 256, 0, stream>>>(x1, qb);
    k_convert<<<convBlocks, 256, 0, stream>>>(x2, kb);

    dim3 gScores(C / 128, C / 128, BATCH);           // 4 x 4 x 16
    k_scores<<<gScores, 256, 0, stream>>>(qb, kb, S);

    k_softmax_row<<<BATCH * C, 256, 0, stream>>>(S, Pa);
    k_softmax_col<<<BATCH * C, 256, 0, stream>>>(S, Pb);

    dim3 gOut(D / 128, C / 128, BATCH);              // 32 x 4 x 16
    const size_t half = NE;
    k_out<<<gOut, 256, 0, stream>>>(Pa, kb, out);         // out_a = Pa . k
    k_out<<<gOut, 256, 0, stream>>>(Pb, qb, out + half);  // out_b = Pb . q
}
